// MambaBlock_28698971472569
// MI455X (gfx1250) — compile-verified, matrix operand order checked by reading, not run on hardware
//
#include <hip/hip_runtime.h>
#include <hip/hip_bf16.h>

// Problem constants (from reference)
#define L_SEQ   1024
#define BATCH   2
#define DMODEL  2048
#define DINNER  4096
#define NSTATE  16
#define DCONV   4
#define DTRANK  128
#define TOKENS  (L_SEQ * BATCH)        // token t = l*BATCH + b
#define XDBL_W  (DTRANK + 2 * NSTATE)  // 160

typedef _Float16 v16h __attribute__((ext_vector_type(16)));
typedef _Float16 v8h  __attribute__((ext_vector_type(8)));
typedef float    v8f  __attribute__((ext_vector_type(8)));

#define LDS_STRIDE 40   // 32 halves + 8 pad (80B rows, 16B aligned)

// ---- 16 contiguous K-elements of one row -> two v8h (f16) ----
__device__ __forceinline__ void load_row16(const float* gp, v8h& lo, v8h& hi) {
  float4 a0 = ((const float4*)gp)[0];
  float4 a1 = ((const float4*)gp)[1];
  float4 a2 = ((const float4*)gp)[2];
  float4 a3 = ((const float4*)gp)[3];
  lo[0]=(_Float16)a0.x; lo[1]=(_Float16)a0.y; lo[2]=(_Float16)a0.z; lo[3]=(_Float16)a0.w;
  lo[4]=(_Float16)a1.x; lo[5]=(_Float16)a1.y; lo[6]=(_Float16)a1.z; lo[7]=(_Float16)a1.w;
  hi[0]=(_Float16)a2.x; hi[1]=(_Float16)a2.y; hi[2]=(_Float16)a2.z; hi[3]=(_Float16)a2.w;
  hi[4]=(_Float16)a3.x; hi[5]=(_Float16)a3.y; hi[6]=(_Float16)a3.z; hi[7]=(_Float16)a3.w;
}
__device__ __forceinline__ void load_row16(const _Float16* gp, v8h& lo, v8h& hi) {
  lo = *(const v8h*)gp;
  hi = *(const v8h*)(gp + 8);
}

// WMMA fragment (16 rows x 32 K, f16) from padded LDS tile.
// Lane r=lane&15 owns row r; kBase=(lane&16)?8:0.
// elems 0..7 -> K=kBase..kBase+7 ; elems 8..15 -> K=16+kBase..23+kBase (2x b128)
__device__ __forceinline__ v16h frag_from_lds(const _Float16* s, int row, int kBase) {
  const _Float16* p = s + row * LDS_STRIDE + kBase;
  v8h lo = *(const v8h*)(p);
  v8h hi = *(const v8h*)(p + 16);
  return __builtin_shufflevector(lo, hi, 0,1,2,3,4,5,6,7,8,9,10,11,12,13,14,15);
}

// C[M x Nn] = A[M x Kk] * Bw[Nn x Kk]^T
// TA: f32 or f16 A-operand (f16 in LDS either way). TC: f32 or f16 output.
// EPI==1: C = softplus(C + bias[col]).
// Block: 256 threads = 8 waves. Tile 128M x 128N, double-buffered LDS.
// All 8 B fragments are loaded into registers up front so the 16 ds_load_b128
// pipeline through the DS unit instead of one-load-one-wait per WMMA.
template<typename TA, typename TC, int EPI>
__global__ __launch_bounds__(256)
void gemm_wmma(const TA* __restrict__ A, int lda,
               const float* __restrict__ Bw, int ldb,
               TC* __restrict__ C, int ldc,
               int Nn, int Kk, const float* __restrict__ bias) {
  __shared__ _Float16 As[2][128 * LDS_STRIDE];   // 2 x 10 KB
  __shared__ _Float16 Bs[2][128 * LDS_STRIDE];   // 2 x 10 KB

  const int tid  = threadIdx.x;
  const int lane = tid & 31;
  const int w    = tid >> 5;          // wave 0..7 -> 16-row M strip
  const int m0   = blockIdx.x * 128;  // M = 2048 (multiple of 128): no row guard
  const int n0   = blockIdx.y * 128;

  v8f acc[8] = {};

  const int r   = tid >> 1;           // staging row 0..127 (2 threads per row)
  const int seg = (tid & 1) * 16;     // 16 K-elements each
  const bool bvalid = (n0 + r) < Nn;  // x_proj: N=160 guard

  v8h ra0, ra1, rb0, rb1;
  auto stage_loads = [&](int k0) {
    load_row16(A + (size_t)(m0 + r) * lda + k0 + seg, ra0, ra1);
    if (bvalid) {
      load_row16(Bw + (size_t)(n0 + r) * ldb + k0 + seg, rb0, rb1);
    } else {
      rb0 = (v8h)(_Float16)0.f; rb1 = (v8h)(_Float16)0.f;
    }
  };
  auto stage_store = [&](int buf) {
    _Float16* sa = &As[buf][r * LDS_STRIDE + seg];
    *(v8h*)sa = ra0; *(v8h*)(sa + 8) = ra1;       // 2x ds_store_b128
    _Float16* sb = &Bs[buf][r * LDS_STRIDE + seg];
    *(v8h*)sb = rb0; *(v8h*)(sb + 8) = rb1;
  };

  stage_loads(0);
  stage_store(0);
  __syncthreads();

  int cur = 0;
  const int kBase = (lane & 16) ? 8 : 0;
  const int fr    = lane & 15;
  for (int k0 = 0; k0 < Kk; k0 += 32) {
    const bool has_next = (k0 + 32) < Kk;
    if (has_next) {
      stage_loads(k0 + 32);           // global loads overlap the WMMAs below
      __builtin_prefetch(A + (size_t)(m0 + r) * lda + k0 + 64 + seg, 0, 1);
    }
    // issue ALL fragment loads first (2 + 16 ds_load_b128 in flight) ...
    const v16h af = frag_from_lds(As[cur], 16 * w + fr, kBase);
    v16h bf[8];
#pragma unroll
    for (int nt = 0; nt < 8; ++nt)
      bf[nt] = frag_from_lds(Bs[cur], nt * 16 + fr, kBase);
    // ... then drain the WMMA chain
#pragma unroll
    for (int nt = 0; nt < 8; ++nt)
      acc[nt] = __builtin_amdgcn_wmma_f32_16x16x32_f16(
          false, af, false, bf[nt], (short)0, acc[nt], false, false);
    if (has_next) {
      stage_store(cur ^ 1);
      __syncthreads();                // single barrier per K-step
      cur ^= 1;
    }
  }

  // epilogue: lanes 0-15 -> rows i, lanes 16-31 -> rows 8+i
  const int row0 = m0 + 16 * w + ((lane & 16) ? 8 : 0);
#pragma unroll
  for (int nt = 0; nt < 8; ++nt) {
    const int col = n0 + nt * 16 + fr;
    if (col < Nn) {
#pragma unroll
      for (int i = 0; i < 8; ++i) {
        float v = acc[nt][i];
        if (EPI == 1) {
          v += bias[col];
          v = (v > 20.f) ? v : log1pf(__expf(v));   // softplus
        }
        C[(size_t)(row0 + i) * ldc + col] = (TC)v;
      }
    }
  }
}

// causal depthwise conv1d(4) + bias + SiLU; f16 in/out, f32 math
__global__ __launch_bounds__(256)
void conv_silu(const _Float16* __restrict__ xz, const float* __restrict__ cw,
               const float* __restrict__ cb, _Float16* __restrict__ xc) {
  const int idx = blockIdx.x * 256 + threadIdx.x;  // = t*DINNER + d
  const int d = idx & (DINNER - 1);
  const int t = idx >> 12;
  const int b = t & (BATCH - 1);
  const int l = t >> 1;
  float s = cb[d];
#pragma unroll
  for (int j = 0; j < DCONV; ++j) {
    const int ll = l - (DCONV - 1) + j;
    if (ll >= 0)
      s += (float)xz[(size_t)(ll * BATCH + b) * (2 * DINNER) + d] * cw[d * DCONV + j];
  }
  xc[idx] = (_Float16)(s / (1.f + __expf(-s)));    // SiLU
}

// Selective scan: one lane per (b, d, n); 16-lane butterfly reduces over n.
// Block = 256 threads = 16 channels (same b). y aliases xc (f16).
__global__ __launch_bounds__(256)
void scan_kernel(const _Float16* __restrict__ delta, const _Float16* __restrict__ xc,
                 const _Float16* __restrict__ xz,   const _Float16* __restrict__ xdbl,
                 const float* __restrict__ A_log,   const float* __restrict__ Dp,
                 _Float16* __restrict__ y) {
  __shared__ float sbuf[80];   // [0:16) dlt, [16:32) x, [32:48) z, [48:80) B|C
  const int tid = threadIdx.x;
  const int ch  = tid >> 4;                 // channel within block, 0..15
  const int n   = tid & 15;                 // state index
  const int g   = blockIdx.x * 16 + ch;     // global (b,d) channel
  const int b   = g >> 12;                  // DINNER channels per batch
  const int d   = g & (DINNER - 1);
  const int d0  = (blockIdx.x * 16) & (DINNER - 1);

  const float a  = -__expf(A_log[d * NSTATE + n]);
  const float Dv = Dp[d];
  float h = 0.f;

  for (int l = 0; l < L_SEQ; ++l) {
    const int t = l * BATCH + b;
    if (tid < 80) {
      float v;
      if      (tid < 16) v = (float)delta[(size_t)t * DINNER + d0 + tid];
      else if (tid < 32) v = (float)xc   [(size_t)t * DINNER + d0 + (tid - 16)];
      else if (tid < 48) v = (float)xz   [(size_t)t * (2 * DINNER) + DINNER + d0 + (tid - 32)];
      else               v = (float)xdbl [(size_t)t * XDBL_W + DTRANK + (tid - 48)];
      sbuf[tid] = v;
    }
    __syncthreads();
    const float dlt = sbuf[ch];
    const float xv  = sbuf[16 + ch];
    const float zv  = sbuf[32 + ch];
    const float Bn  = sbuf[48 + n];
    const float Cn  = sbuf[64 + n];

    h = __expf(dlt * a) * h + (dlt * xv) * Bn;       // h = dA*h + dB*u
    float red = h * Cn;
#pragma unroll
    for (int off = 8; off; off >>= 1) red += __shfl_xor(red, off, 16);
    if (n == 0) {
      float yv = red + xv * Dv;
      yv *= zv / (1.f + __expf(-zv));                // * silu(z)
      y[(size_t)t * DINNER + d] = (_Float16)yv;
    }
    __syncthreads();
  }
}

extern "C" void kernel_launch(void* const* d_in, const int* in_sizes, int n_in,
                              void* d_out, int out_size, void* d_ws, size_t ws_size,
                              hipStream_t stream) {
  (void)in_sizes; (void)n_in; (void)out_size; (void)ws_size;
  const float* hidden   = (const float*)d_in[0];  // [L,B,DMODEL] = [t][DMODEL]
  const float* in_w     = (const float*)d_in[1];  // [2*DINNER][DMODEL]
  const float* conv_w   = (const float*)d_in[2];  // [DINNER][DCONV]
  const float* conv_b   = (const float*)d_in[3];  // [DINNER]
  const float* xproj_w  = (const float*)d_in[4];  // [160][DINNER]
  const float* dtproj_w = (const float*)d_in[5];  // [DINNER][DTRANK]
  const float* dtproj_b = (const float*)d_in[6];  // [DINNER]
  const float* A_log    = (const float*)d_in[7];  // [DINNER][NSTATE]
  const float* Dp       = (const float*)d_in[8];  // [DINNER]
  const float* outw     = (const float*)d_in[9];  // [DMODEL][DINNER]
  float* out = (float*)d_out;                     // [t][DMODEL]

  // f16 intermediates: ~68 MB total
  _Float16* wsh   = (_Float16*)d_ws;
  _Float16* xz    = wsh;                               // [t][2*DINNER]
  _Float16* xc    = xz   + (size_t)TOKENS * 2 * DINNER;// [t][DINNER] (also y)
  _Float16* xdbl  = xc   + (size_t)TOKENS * DINNER;    // [t][160]
  _Float16* delta = xdbl + (size_t)TOKENS * XDBL_W;    // [t][DINNER]

  const dim3 blk(256);
  // 1) in_proj: xz[t][0:8192]  (M=2048, N=8192, K=2048)
  gemm_wmma<float, _Float16, 0><<<dim3(TOKENS / 128, (2 * DINNER) / 128), blk, 0, stream>>>(
      hidden, DMODEL, in_w, DMODEL, xz, 2 * DINNER, 2 * DINNER, DMODEL, nullptr);
  // 2) causal conv + SiLU -> xc
  conv_silu<<<(TOKENS * DINNER) / 256, 256, 0, stream>>>(xz, conv_w, conv_b, xc);
  // 3) x_proj: xdbl[t][0:160]  (N=160, guarded; grid.y = ceil(160/128) = 2)
  gemm_wmma<_Float16, _Float16, 0><<<dim3(TOKENS / 128, 2), blk, 0, stream>>>(
      xc, DINNER, xproj_w, DINNER, xdbl, XDBL_W, XDBL_W, DINNER, nullptr);
  // 4) dt_proj + softplus -> delta  (K=128)
  gemm_wmma<_Float16, _Float16, 1><<<dim3(TOKENS / 128, DINNER / 128), blk, 0, stream>>>(
      xdbl, XDBL_W, dtproj_w, DTRANK, delta, DINNER, DINNER, DTRANK, dtproj_b);
  // 5) selective scan (y aliases xc)
  scan_kernel<<<(BATCH * DINNER) / 16, 256, 0, stream>>>(
      delta, xc, xz, xdbl, A_log, Dp, xc);
  // 6) out_proj -> d_out (f32)
  gemm_wmma<_Float16, float, 0><<<dim3(TOKENS / 128, DMODEL / 128), blk, 0, stream>>>(
      xc, DINNER, outw, DINNER, out, DMODEL, DMODEL, DINNER, nullptr);
}
